// ScoreMechain_46505905881424
// MI455X (gfx1250) — matrix operand order repaired and checked by reading, not compiled, not run on hardware
//
#include <hip/hip_runtime.h>

typedef __attribute__((ext_vector_type(16))) _Float16     v16h;
typedef __attribute__((ext_vector_type(2)))  _Float16     h2;
typedef __attribute__((ext_vector_type(8)))  float        v8f;
typedef __attribute__((ext_vector_type(8)))  unsigned int v8u;

#define JN    17
#define HD    32
#define NVIS  9
#define NOCC  8
#define SLOPE 0.01f
#define SPB   16      // samples per block (= WMMA M)
#define NTHREADS 128  // 4 waves

// ---- weight-fragment tile ids in d_ws (each tile: 32 lanes x 16 halves = 1KB)
#define T_UP   0    // 68 tiles: t = nt*2 + kt   (W_up 34x544, K padded to 64)
#define T_UD   68   // 4 tiles:  t = 68 + kt*2 + nt (W_ud 64x32)
#define T_Q    72   // 2 tiles
#define T_K    74   // 2 tiles
#define T_V    76   // 2 tiles
#define T_D1   78   // 2 tiles
#define T_D2   80   // 2 tiles
#define NTILES 82
#define WS_NEED ((size_t)NTILES * 512 * sizeof(_Float16))

// A-fragment (16-bit, 16x32) element->K map per CDNA5 ISA 7.12.2:
// lanes 0-15 hold K in {0..7,16..23}; lanes 16-31 hold K in {8..15,24..31}.
__device__ __forceinline__ int akidx(int lane, int e) {
  return (e & 7) | (((lane >> 4) & 1) << 3) | ((e >> 3) << 4);
}

union FragU { v16h h; v8u u; };

// Build A fragment from an LDS row of 32 f16 (K-pairs are contiguous even addrs).
__device__ __forceinline__ v16h lds_afrag(const _Float16* row, int lane) {
  FragU f;
#pragma unroll
  for (int r = 0; r < 8; ++r) {
    const int k = akidx(lane, 2 * r);  // even
    f.u[r] = *(const unsigned int*)(row + k);
  }
  return f.h;
}

// B-fragment (16-bit, 32x16): lane = N + 16*h, element e holds K = akidx(lane, e) (same as A).
__device__ __forceinline__ v16h bfrag_f32(const float* __restrict__ W, int K, int N,
                                          int kbase, int nbase, int lane) {
  const int n  = nbase + (lane & 15);
  const int kb = kbase;
  v16h b;
#pragma unroll
  for (int e = 0; e < 16; ++e) {
    const int k = kb + akidx(lane, e);
    b[e] = (k < K) ? (_Float16)W[(size_t)k * N + n] : (_Float16)0.f;
  }
  return b;
}

template <bool USEWS>
__device__ __forceinline__ v16h get_bfrag(const _Float16* __restrict__ wsf, int tile,
                                          const float* __restrict__ W, int K, int N,
                                          int kbase, int nbase, int lane) {
  if (USEWS) return *(const v16h*)(wsf + (size_t)tile * 512 + lane * 16);
  return bfrag_f32(W, K, N, kbase, nbase, lane);
}

__device__ __forceinline__ float leakyf(float v) { return fmaxf(v, SLOPE * v); }

__device__ __forceinline__ v8f wmma16(v16h a, v16h b, v8f c) {
  return __builtin_amdgcn_wmma_f32_16x16x32_f16(false, a, false, b, (short)0, c,
                                                false, false);
}

// ---------------- prep: convert all weights to fragment-linear f16 in d_ws ----
__global__ __launch_bounds__(32) void prep_frags(
    const float* __restrict__ W_up, const float* __restrict__ W_ud,
    const float* __restrict__ W_q,  const float* __restrict__ W_k,
    const float* __restrict__ W_v,  const float* __restrict__ W_d1,
    const float* __restrict__ W_d2, _Float16* __restrict__ wsf) {
  const int t = blockIdx.x, lane = threadIdx.x;
  const float* W; int K, N, kb, nb;
  if (t < T_UD)      { W = W_up; K = 34; N = JN * HD; kb = (t & 1) * 32; nb = (t >> 1) * 16; }
  else if (t < T_Q)  { int q = t - T_UD; W = W_ud; K = 64; N = HD; kb = (q >> 1) * 32; nb = (q & 1) * 16; }
  else if (t < T_K)  { W = W_q;  K = HD; N = HD; kb = 0; nb = (t - T_Q)  * 16; }
  else if (t < T_V)  { W = W_k;  K = HD; N = HD; kb = 0; nb = (t - T_K)  * 16; }
  else if (t < T_D1) { W = W_v;  K = HD; N = HD; kb = 0; nb = (t - T_V)  * 16; }
  else if (t < T_D2) { W = W_d1; K = HD; N = HD; kb = 0; nb = (t - T_D1) * 16; }
  else               { W = W_d2; K = HD; N = HD; kb = 0; nb = (t - T_D2) * 16; }
  *(v16h*)(wsf + (size_t)t * 512 + lane * 16) = bfrag_f32(W, K, N, kb, nb, lane);
}

// ---------------- fused forward -----------------------------------------------
template <bool USEWS>
__global__ __launch_bounds__(NTHREADS) void score_fused(
    const float* __restrict__ x, const unsigned char* __restrict__ msk,
    const float* __restrict__ pos,
    const float* __restrict__ W_up, const float* __restrict__ b_up,
    const float* __restrict__ W_ud, const float* __restrict__ b_ud,
    const float* __restrict__ W_q,  const float* __restrict__ b_q,
    const float* __restrict__ W_k,  const float* __restrict__ b_k,
    const float* __restrict__ W_v,  const float* __restrict__ b_v,
    const float* __restrict__ W_d1, const float* __restrict__ b_d1,
    const float* __restrict__ W_d2, const float* __restrict__ b_d2,
    const float* __restrict__ W_s,  const float* __restrict__ b_s,
    const _Float16* __restrict__ wsf, float* __restrict__ out, int B)
{
  __shared__ _Float16 s_up[SPB][JN][HD];    // stage1 out; reused for K in stage3
  __shared__ _Float16 s_ux[SPB][JN][HD];    // stage2 out; overwritten by V in stage3
  __shared__ _Float16 s_q [SPB][NOCC][HD];  // q; overwritten by attention output o
  __shared__ _Float16 s_x2[SPB][NOCC][HD];  // residual
  __shared__ float    s_pos[JN][HD];
  __shared__ int      s_idx[SPB][JN];       // slot -> joint (0..8 visible, 9..16 occluded)
  __shared__ float    s_scr[4][SPB * HD];   // per-wave scratch (t f16 / o2 f32)

  const int tid  = threadIdx.x;
  const int lane = tid & 31;
  const int w    = tid >> 5;
  const int s0   = blockIdx.x * SPB;
  const int m    = lane & 15;
  const int gh   = lane >> 4;
  const int n    = lane & 15;

  // ---------------- stage 0 ---------------------------------------------------
  for (int i = tid; i < JN * HD; i += NTHREADS) s_pos[i / HD][i % HD] = pos[i];
  if (tid < SPB) {
    int sg = s0 + tid; if (sg > B - 1) sg = B - 1;
    const unsigned char* mr = msk + (size_t)sg * JN;
    int ct = 0, cf = 0;
    for (int j = 0; j < JN; ++j) {
      if (mr[j]) s_idx[tid][ct++] = j;
      else       s_idx[tid][NVIS + (cf++)] = j;
    }
  }
  if (s0 + SPB < B)  // warm L2 for the next tile of x (global_prefetch_b8)
    __builtin_prefetch(x + (size_t)(s0 + SPB) * (2 * JN) + tid * 4, 0, 1);
  __syncthreads();

  // ---------------- stage 1: up = leaky(x @ W_up + b_up) ----------------------
  {
    int sg = s0 + m; if (sg > B - 1) sg = B - 1;
    const float* xr = x + (size_t)sg * (2 * JN);
    v16h a0, a1;
#pragma unroll
    for (int r = 0; r < 8; ++r) {               // K pairs contiguous, 8B aligned
      const int k = akidx(lane, 2 * r);
      const float2 p = *(const float2*)(xr + k);
      a0[2 * r] = (_Float16)p.x; a0[2 * r + 1] = (_Float16)p.y;
    }
#pragma unroll
    for (int e = 0; e < 16; ++e) {
      const int k = 32 + akidx(lane, e);
      a1[e] = (k < 34) ? (_Float16)xr[k] : (_Float16)0.f;
    }
    for (int nt = w; nt < 34; nt += 4) {
      v16h b0 = get_bfrag<USEWS>(wsf, T_UP + 2 * nt + 0, W_up, 34, JN * HD, 0,  nt * 16, lane);
      v16h b1 = get_bfrag<USEWS>(wsf, T_UP + 2 * nt + 1, W_up, 34, JN * HD, 32, nt * 16, lane);
      const float bv = b_up[nt * 16 + n];
      v8f c;
#pragma unroll
      for (int r = 0; r < 8; ++r) c[r] = bv;
      c = wmma16(a0, b0, c);
      c = wmma16(a1, b1, c);
      const int f = nt * 16 + n;
#pragma unroll
      for (int r = 0; r < 8; ++r)
        s_up[r + 8 * gh][f >> 5][f & 31] = (_Float16)leakyf(c[r]);
    }
  }
  __syncthreads();

  // ---------------- stage 2: gather+concat(pos) -> u = leaky(. @ W_ud) --------
  for (int i = tid; i < SPB * NOCC * (HD / 2); i += NTHREADS) {  // x2 copy, b32 wide
    const int s  = i / (NOCC * (HD / 2));
    const int jq = (i / (HD / 2)) % NOCC;
    const int hp = (i % (HD / 2)) * 2;
    *(unsigned int*)&s_x2[s][jq][hp] =
        *(const unsigned int*)&s_up[s][s_idx[s][NVIS + jq]][hp];
  }
  {
    v16h bud[2][2];
#pragma unroll
    for (int kt = 0; kt < 2; ++kt)
#pragma unroll
      for (int nt = 0; nt < 2; ++nt)
        bud[kt][nt] = get_bfrag<USEWS>(wsf, T_UD + kt * 2 + nt, W_ud, 64, HD,
                                       kt * 32, nt * 16, lane);
    for (int j = w; j < JN; j += 4) {
      const int gj = s_idx[m][j];
      v16h a0 = lds_afrag(&s_up[m][gj][0], lane);
      v16h a1;
#pragma unroll
      for (int e = 0; e < 16; ++e) a1[e] = (_Float16)s_pos[gj][akidx(lane, e)];
#pragma unroll
      for (int nt = 0; nt < 2; ++nt) {
        const float bv = b_ud[nt * 16 + n];
        v8f c;
#pragma unroll
        for (int r = 0; r < 8; ++r) c[r] = bv;
        c = wmma16(a0, bud[0][nt], c);
        c = wmma16(a1, bud[1][nt], c);
#pragma unroll
        for (int r = 0; r < 8; ++r)
          s_ux[r + 8 * gh][j][nt * 16 + n] = (_Float16)leakyf(c[r]);
      }
    }
  }
  __syncthreads();

  // ---------------- stage 3: q / k / v (k -> s_up, v -> s_ux in place) --------
  {
    v16h bq[2], bk[2], bv2[2];
#pragma unroll
    for (int nt = 0; nt < 2; ++nt) {
      bq[nt]  = get_bfrag<USEWS>(wsf, T_Q + nt, W_q, HD, HD, 0, nt * 16, lane);
      bk[nt]  = get_bfrag<USEWS>(wsf, T_K + nt, W_k, HD, HD, 0, nt * 16, lane);
      bv2[nt] = get_bfrag<USEWS>(wsf, T_V + nt, W_v, HD, HD, 0, nt * 16, lane);
    }
    for (int j = w; j < JN; j += 4) {
      v16h a = lds_afrag(&s_ux[m][j][0], lane);
#pragma unroll
      for (int nt = 0; nt < 2; ++nt) {
        float bb = b_k[nt * 16 + n];
        v8f ck;
#pragma unroll
        for (int r = 0; r < 8; ++r) ck[r] = bb;
        ck = wmma16(a, bk[nt], ck);
#pragma unroll
        for (int r = 0; r < 8; ++r)
          s_up[r + 8 * gh][j][nt * 16 + n] = (_Float16)ck[r];

        bb = b_v[nt * 16 + n];
        v8f cv;
#pragma unroll
        for (int r = 0; r < 8; ++r) cv[r] = bb;
        cv = wmma16(a, bv2[nt], cv);
#pragma unroll
        for (int r = 0; r < 8; ++r)
          s_ux[r + 8 * gh][j][nt * 16 + n] = (_Float16)cv[r];

        if (j >= NVIS) {  // wave-uniform, EXEC stays full for WMMA
          bb = b_q[nt * 16 + n];
          v8f cq;
#pragma unroll
          for (int r = 0; r < 8; ++r) cq[r] = bb;
          cq = wmma16(a, bq[nt], cq);
#pragma unroll
          for (int r = 0; r < 8; ++r)
            s_q[r + 8 * gh][j - NVIS][nt * 16 + n] = (_Float16)cq[r];
        }
      }
    }
  }
  __syncthreads();

  // ---------------- stage 4: attention, packed f16 VALU -----------------------
  {
    const int s  = tid >> 3;
    const int qj = tid & 7;
    const h2* qp = (const h2*)&s_q[s][qj][0];
    h2 qv[16];
#pragma unroll
    for (int i = 0; i < 16; ++i) qv[i] = qp[i];

    float att[JN];
    float mx = -1e30f;
#pragma unroll
    for (int kj = 0; kj < JN; ++kj) {
      const h2* kp = (const h2*)&s_up[s][kj][0];
      h2 acc = {(_Float16)0.f, (_Float16)0.f};
#pragma unroll
      for (int i = 0; i < 16; ++i) acc += qv[i] * kp[i];   // v_pk_fma_f16
      float d = ((float)acc[0] + (float)acc[1]) * 0.17677669529663689f;
      att[kj] = d;
      mx = fmaxf(mx, d);
    }
    float ssum = 0.f;
#pragma unroll
    for (int kj = 0; kj < JN; ++kj) { att[kj] = __expf(att[kj] - mx); ssum += att[kj]; }
    const float inv = 1.f / ssum;

    h2 ov[16];
#pragma unroll
    for (int i = 0; i < 16; ++i) ov[i] = (h2){(_Float16)0.f, (_Float16)0.f};
#pragma unroll
    for (int kj = 0; kj < JN; ++kj) {
      const _Float16 wh = (_Float16)(att[kj] * inv);
      const h2 w2 = {wh, wh};
      const h2* vp = (const h2*)&s_ux[s][kj][0];
#pragma unroll
      for (int i = 0; i < 16; ++i) ov[i] += w2 * vp[i];
    }
    h2* op = (h2*)&s_q[s][qj][0];  // own slot only: safe in-place over q
#pragma unroll
    for (int i = 0; i < 16; ++i) op[i] = ov[i];
  }
  __syncthreads();

  // ---------------- stage 5: leaky(o@W_d1)@W_d2 + x2 -> sigmoid(.@W_s) --------
  {
    v16h bd1[2], bd2[2];
#pragma unroll
    for (int nt = 0; nt < 2; ++nt) {
      bd1[nt] = get_bfrag<USEWS>(wsf, T_D1 + nt, W_d1, HD, HD, 0, nt * 16, lane);
      bd2[nt] = get_bfrag<USEWS>(wsf, T_D2 + nt, W_d2, HD, HD, 0, nt * 16, lane);
    }
    _Float16* tbuf = (_Float16*)&s_scr[w][0];
    float*    obuf = &s_scr[w][0];

    for (int jq = w; jq < NOCC; jq += 4) {
      v16h a = lds_afrag(&s_q[m][jq][0], lane);
#pragma unroll
      for (int nt = 0; nt < 2; ++nt) {
        const float bv = b_d1[nt * 16 + n];
        v8f c;
#pragma unroll
        for (int r = 0; r < 8; ++r) c[r] = bv;
        c = wmma16(a, bd1[nt], c);
#pragma unroll
        for (int r = 0; r < 8; ++r)
          tbuf[(r + 8 * gh) * HD + nt * 16 + n] = (_Float16)leakyf(c[r]);
      }
      v16h a2 = lds_afrag(&tbuf[m * HD], lane);
#pragma unroll
      for (int nt = 0; nt < 2; ++nt) {
        const float bv = b_d2[nt * 16 + n];
        v8f c2;
#pragma unroll
        for (int r = 0; r < 8; ++r)
          c2[r] = bv + (float)s_x2[r + 8 * gh][jq][nt * 16 + n];
        c2 = wmma16(a2, bd2[nt], c2);
#pragma unroll
        for (int r = 0; r < 8; ++r)
          obuf[(r + 8 * gh) * HD + nt * 16 + n] = c2[r];
      }
      if (lane < 16 && (s0 + lane) < B) {
        float acc = b_s[0];
#pragma unroll
        for (int h = 0; h < HD; ++h) acc += obuf[lane * HD + h] * W_s[h];
        out[(size_t)(s0 + lane) * NOCC + jq] = 1.f / (1.f + __expf(-acc));
      }
    }
  }
}

extern "C" void kernel_launch(void* const* d_in, const int* in_sizes, int n_in,
                              void* d_out, int out_size, void* d_ws, size_t ws_size,
                              hipStream_t stream) {
  (void)n_in; (void)out_size;
  const float*         x    = (const float*)d_in[0];
  const unsigned char* m    = (const unsigned char*)d_in[1];
  const float*         pos  = (const float*)d_in[2];
  const float* W_up = (const float*)d_in[3];
  const float* b_up = (const float*)d_in[4];
  const float* W_ud = (const float*)d_in[5];
  const float* b_ud = (const float*)d_in[6];
  const float* W_q  = (const float*)d_in[7];
  const float* b_q  = (const float*)d_in[8];
  const float* W_k  = (const float*)d_in[9];
  const float* b_k  = (const float*)d_in[10];
  const float* W_v  = (const float*)d_in[11];
  const float* b_v  = (const float*)d_in[12];
  const float* W_d1 = (const float*)d_in[13];
  const float* b_d1 = (const float*)d_in[14];
  const float* W_d2 = (const float*)d_in[15];
  const float* b_d2 = (const float*)d_in[16];
  const float* W_s  = (const float*)d_in[17];
  const float* b_s  = (const float*)d_in[18];
  float* out = (float*)d_out;

  const int B = in_sizes[0] / (2 * JN);
  const int blocks = (B + SPB - 1) / SPB;

  if (ws_size >= WS_NEED) {
    _Float16* wsf = (_Float16*)d_ws;
    prep_frags<<<NTILES, 32, 0, stream>>>(W_up, W_ud, W_q, W_k, W_v, W_d1, W_d2, wsf);
    score_fused<true><<<blocks, NTHREADS, 0, stream>>>(
        x, m, pos, W_up, b_up, W_ud, b_ud, W_q, b_q, W_k, b_k, W_v, b_v,
        W_d1, b_d1, W_d2, b_d2, W_s, b_s, wsf, out, B);
  } else {
    score_fused<false><<<blocks, NTHREADS, 0, stream>>>(
        x, m, pos, W_up, b_up, W_ud, b_ud, W_q, b_q, W_k, b_k, W_v, b_v,
        W_d1, b_d1, W_d2, b_d2, W_s, b_s, (const _Float16*)nullptr, out, B);
  }
}
